// MultiheadAttention_9096740733073
// MI455X (gfx1250) — matrix operand order repaired and checked by reading, not compiled, not run on hardware
//
#include <hip/hip_runtime.h>
#include <hip/hip_bf16.h>

#define B_ 4
#define L_ 2048
#define E_ 1024
#define H_ 16
#define D_ 64
#define HUGE_MASK 1e31f

typedef __attribute__((ext_vector_type(16))) __bf16 v16bf;
typedef __attribute__((ext_vector_type(8)))  float  v8f;
typedef __attribute__((ext_vector_type(4)))  int    v4i;
typedef __attribute__((address_space(1)))    v4i    gv4i; // global int4
typedef __attribute__((address_space(3)))    v4i    lv4i; // LDS int4

#if __has_builtin(__builtin_amdgcn_global_load_async_to_lds_b128)
#define HAS_ASYNC 1
#else
#define HAS_ASYNC 0
#endif

// 16-byte global -> LDS copy. Async (ASYNCcnt-tracked) when available.
__device__ __forceinline__ void cp_b128(void* lds, const void* gsrc) {
#if HAS_ASYNC
  __builtin_amdgcn_global_load_async_to_lds_b128(
      (gv4i*)(size_t)gsrc,
      (lv4i*)(unsigned)(size_t)lds, // generic LDS addr: offset in low 32 bits
      /*offset=*/0, /*cpol=*/0);
#else
  *(uint4*)lds = *(const uint4*)gsrc;
#endif
}

__device__ __forceinline__ void wait_async0() {
#if __has_builtin(__builtin_amdgcn_s_wait_asynccnt)
  __builtin_amdgcn_s_wait_asynccnt(0);
#elif HAS_ASYNC
  asm volatile("s_wait_asynccnt 0x0" ::: "memory");
#endif
}

__device__ __forceinline__ v8f wmma_bf16(v16bf a, v16bf b, v8f c) {
  // D = A(16x32 bf16) * B(32x16 bf16) + C(16x16 f32)
  return __builtin_amdgcn_wmma_f32_16x16x32_bf16(
      /*neg_a=*/false, a, /*neg_b=*/false, b,
      /*c_mod=*/(short)0, c, /*reuse_a=*/false, /*reuse_b=*/false);
}

// A-fragment: 16x32 bf16 from a row-major LDS tile (leading dim `ld` elems).
// Lane m in [0,16): rows; half selects K groups {0-7,16-23} vs {8-15,24-31}.
__device__ __forceinline__ v16bf load_A(const __bf16* p, int row_base, int k_base, int ld) {
  const int lane = threadIdx.x & 31;
  const int half = lane >> 4;
  const int m = lane & 15;
  const __bf16* rowp = p + (size_t)(row_base + m) * ld + k_base + 8 * half;
  v16bf a;
#pragma unroll
  for (int j = 0; j < 8; ++j) {
    const int k0 = 2 * j + (j >= 4 ? 8 : 0);
    a[2 * j]     = rowp[k0];
    a[2 * j + 1] = rowp[k0 + 1];
  }
  return a;
}

// B-fragment (K=32 x N=16) read from a B-transposed LDS tile laid out [N][K]
// (row-major, leading dim `ld`): lane n = column, half selects K groups {0-7,16-23} vs {8-15,24-31}.
__device__ __forceinline__ v16bf load_B(const __bf16* p, int n_base, int k_base, int ld) {
  const int lane = threadIdx.x & 31;
  const int n = lane & 15;
  const int kb = (lane >> 4) * 8;
  const __bf16* rowp = p + (size_t)(n_base + n) * ld + k_base + kb;
  v16bf b;
#pragma unroll
  for (int j = 0; j < 16; ++j) b[j] = rowp[j + (j >= 8 ? 8 : 0)];
  return b;
}

#define PROJ_ROWS 256 // 32 rows per wave (2 M-tiles), 8 waves
#define KC 32

// Out[bh][L][D] (transposed==0) or Out[bh][D][L] (transposed==1), bf16.
__global__ __launch_bounds__(256)
void proj_kernel(const float* __restrict__ X,    // [B, L, E]
                 const float* __restrict__ W,    // [H, E, D]
                 const float* __restrict__ bias, // [H, D]
                 __bf16* __restrict__ Out,
                 int transposed) {
  __shared__ __bf16 Xs[PROJ_ROWS * KC]; // 16 KB, [row][e]
  __shared__ __bf16 Wt[D_ * KC];        //  4 KB, [d][e]  (== B^T layout)
  const int tid  = threadIdx.x;
  const int wave = tid >> 5;
  const int lane = tid & 31;
  const int bh = blockIdx.y;
  const int b  = bh / H_;
  const int h  = bh % H_;
  const int row0 = blockIdx.x * PROJ_ROWS;
  const float* Xb = X + ((size_t)b * L_ + row0) * E_;
  const float* Wh = W + (size_t)h * E_ * D_;

  v8f acc[2][4] = {};

  for (int e0 = 0; e0 < E_; e0 += KC) {
    __syncthreads();
    for (int idx = tid; idx < PROJ_ROWS * KC; idx += 256) {
      const int r = idx >> 5, e = idx & 31; // consecutive tid -> consecutive e
      Xs[idx] = (__bf16)Xb[(size_t)r * E_ + e0 + e];
    }
    for (int idx = tid; idx < D_ * KC; idx += 256) {
      const int e = idx >> 6, d = idx & 63; // consecutive tid -> consecutive d
      Wt[d * KC + e] = (__bf16)Wh[(size_t)(e0 + e) * D_ + d];
    }
    __syncthreads();
    const v16bf a0 = load_A(Xs, wave * 32, 0, KC);
    const v16bf a1 = load_A(Xs, wave * 32 + 16, 0, KC);
#pragma unroll
    for (int nt = 0; nt < 4; ++nt) {
      const v16bf bB = load_B(Wt, nt * 16, 0, KC); // reused by both M-tiles
      acc[0][nt] = wmma_bf16(a0, bB, acc[0][nt]);
      acc[1][nt] = wmma_bf16(a1, bB, acc[1][nt]);
    }
  }

  const int half = lane >> 4;
  const int n = lane & 15;
#pragma unroll
  for (int mt = 0; mt < 2; ++mt) {
#pragma unroll
    for (int nt = 0; nt < 4; ++nt) {
      const int d = nt * 16 + n;
      const float bv = bias[h * D_ + d];
#pragma unroll
      for (int r = 0; r < 8; ++r) {
        const int l = row0 + wave * 32 + mt * 16 + r + 8 * half;
        const float v = acc[mt][nt][r] + bv;
        const size_t off = transposed ? ((size_t)bh * D_ * L_ + (size_t)d * L_ + l)
                                      : ((size_t)bh * L_ * D_ + (size_t)l * D_ + d);
        Out[off] = (__bf16)v;
      }
    }
  }
}

#define QT 128 // queries per block (16 per wave)
#define KB 64  // keys per step

__global__ __launch_bounds__(256)
void attn_kernel(const __bf16* __restrict__ Q,  // [B*H, L, D]
                 const __bf16* __restrict__ K,  // [B*H, L, D]
                 const __bf16* __restrict__ Vt, // [B*H, D, L]
                 const int*    __restrict__ masks, // [B, L]
                 float* __restrict__ out) {        // [B, L, H*D]
  __shared__ __bf16 Qs[QT * D_];         // 16 KB [qrow][d]
  __shared__ __bf16 Ks[2][KB * D_];      // 2x8 KB [key][d]   (= B^T for Q.K^T)
  __shared__ __bf16 Vs[2][D_ * KB];      // 2x8 KB [d][key]   (= B^T for P.V)
  __shared__ __bf16 Ps[8 * 16 * KB];     // 16 KB per-wave P strips

  const int tid  = threadIdx.x;
  const int wave = tid >> 5;
  const int lane = tid & 31;
  const int half = lane >> 4;
  const int n    = lane & 15;
  const int bh = blockIdx.y;
  const int b  = bh / H_;
  const int h  = bh % H_;
  const int q0 = blockIdx.x * QT;

  const char* Kg  = (const char*)(K + (size_t)bh * L_ * D_);  // + key*D*2 bytes
  const char* Vtg = (const char*)(Vt + (size_t)bh * D_ * L_); // + (d*L + key)*2 bytes

  // Stage Q tile (contiguous 16 KB) + first K/V^T block, all async.
  {
    const char* g = (const char*)(Q + ((size_t)bh * L_ + q0) * D_);
    char* l = (char*)Qs;
    for (int i = tid; i < QT * D_ * 2 / 16; i += 256) cp_b128(l + i * 16, g + i * 16);
  }
  {
    char* l = (char*)Ks[0];
    for (int i = tid; i < KB * D_ * 2 / 16; i += 256) cp_b128(l + i * 16, Kg + i * 16);
    char* lv = (char*)Vs[0];
    for (int i = tid; i < D_ * KB * 2 / 16; i += 256) {
      const int d = i >> 3, c = i & 7; // 8x16B chunks per d-row
      cp_b128(lv + i * 16, Vtg + (size_t)d * L_ * 2 + c * 16);
    }
  }
  wait_async0();
  __syncthreads();

  const v16bf aq0 = load_A(Qs, wave * 16, 0, D_);
  const v16bf aq1 = load_A(Qs, wave * 16, 32, D_);

  float mrow[8], lrow[8];
#pragma unroll
  for (int r = 0; r < 8; ++r) { mrow[r] = -1e30f; lrow[r] = 0.f; }
  v8f o[4] = {};
  const float scale = 0.125f; // 1/sqrt(D)
  __bf16* Pw = Ps + wave * 16 * KB;

  int cur = 0;
  for (int k0 = 0; k0 < L_; k0 += KB, cur ^= 1) {
    // Prefetch next block into the alternate buffers (overlaps with WMMA below).
    if (k0 + KB < L_) {
      const char* gk = Kg + (size_t)(k0 + KB) * D_ * 2;
      char* l = (char*)Ks[cur ^ 1];
      for (int i = tid; i < KB * D_ * 2 / 16; i += 256) cp_b128(l + i * 16, gk + i * 16);
      const char* gv = Vtg + (size_t)(k0 + KB) * 2;
      char* lv = (char*)Vs[cur ^ 1];
      for (int i = tid; i < D_ * KB * 2 / 16; i += 256) {
        const int d = i >> 3, c = i & 7;
        cp_b128(lv + i * 16, gv + (size_t)d * L_ * 2 + c * 16);
      }
    }

    const __bf16* Kc = Ks[cur];
    const __bf16* Vc = Vs[cur];

    // S = scale * Q K^T + mask
    v8f s[4] = {};
#pragma unroll
    for (int nt = 0; nt < 4; ++nt) {
      s[nt] = wmma_bf16(aq0, load_B(Kc, nt * 16, 0, D_), s[nt]);
      s[nt] = wmma_bf16(aq1, load_B(Kc, nt * 16, 32, D_), s[nt]);
    }
#pragma unroll
    for (int nt = 0; nt < 4; ++nt) {
      const int key = k0 + nt * 16 + n;
      const float madd = masks[b * L_ + key] ? 0.f : -HUGE_MASK;
#pragma unroll
      for (int r = 0; r < 8; ++r) s[nt][r] = s[nt][r] * scale + madd;
    }

    // Online softmax: element r holds row (r + 8*half); reduce over 16-lane half.
    float newm[8], alpha[8];
#pragma unroll
    for (int r = 0; r < 8; ++r) {
      float v = fmaxf(fmaxf(s[0][r], s[1][r]), fmaxf(s[2][r], s[3][r]));
#pragma unroll
      for (int off = 1; off < 16; off <<= 1) v = fmaxf(v, __shfl_xor(v, off, 32));
      newm[r] = fmaxf(mrow[r], v);
      alpha[r] = __expf(mrow[r] - newm[r]);
      mrow[r] = newm[r];
    }
    float rsum[8];
#pragma unroll
    for (int r = 0; r < 8; ++r) rsum[r] = 0.f;
#pragma unroll
    for (int nt = 0; nt < 4; ++nt) {
#pragma unroll
      for (int r = 0; r < 8; ++r) {
        const float p = __expf(s[nt][r] - newm[r]);
        rsum[r] += p;
        Pw[(size_t)(r + 8 * half) * KB + nt * 16 + n] = (__bf16)p; // C-layout -> row-major
      }
    }
#pragma unroll
    for (int r = 0; r < 8; ++r) {
      float v = rsum[r];
#pragma unroll
      for (int off = 1; off < 16; off <<= 1) v += __shfl_xor(v, off, 32);
      lrow[r] = lrow[r] * alpha[r] + v;
    }
#pragma unroll
    for (int nt = 0; nt < 4; ++nt)
#pragma unroll
      for (int r = 0; r < 8; ++r) o[nt][r] *= alpha[r];

    // O += P V  (per-wave LDS strip; same-wave DS RAW -> s_wait_dscnt by compiler)
    const v16bf pa0 = load_A(Pw, 0, 0, KB);
    const v16bf pa1 = load_A(Pw, 0, 32, KB);
#pragma unroll
    for (int nt = 0; nt < 4; ++nt) {
      o[nt] = wmma_bf16(pa0, load_B(Vc, nt * 16, 0, KB), o[nt]);
      o[nt] = wmma_bf16(pa1, load_B(Vc, nt * 16, 32, KB), o[nt]);
    }

    wait_async0();   // next block's tiles landed
    __syncthreads(); // all waves done with buf[cur]; buf[cur^1] ready
  }

  // out[b][q][h*D + d]
#pragma unroll
  for (int nt = 0; nt < 4; ++nt) {
#pragma unroll
    for (int r = 0; r < 8; ++r) {
      const int q = q0 + wave * 16 + r + 8 * half;
      const int d = nt * 16 + n;
      out[((size_t)b * L_ + q) * (H_ * D_) + h * D_ + d] = o[nt][r] / lrow[r];
    }
  }
}

extern "C" void kernel_launch(void* const* d_in, const int* in_sizes, int n_in,
                              void* d_out, int out_size, void* d_ws, size_t ws_size,
                              hipStream_t stream) {
  const float* query = (const float*)d_in[0];
  const float* key   = (const float*)d_in[1];
  const float* value = (const float*)d_in[2];
  const int*   masks = (const int*)d_in[3];
  const float* Wq = (const float*)d_in[4];
  const float* bq = (const float*)d_in[5];
  const float* Wk = (const float*)d_in[6];
  const float* bk = (const float*)d_in[7];
  const float* Wv = (const float*)d_in[8];
  const float* bv = (const float*)d_in[9];
  float* out = (float*)d_out;

  const size_t nqkv = (size_t)B_ * H_ * L_ * D_; // 8M bf16 elements per buffer
  __bf16* Qbf = (__bf16*)d_ws;
  __bf16* Kbf = Qbf + nqkv;
  __bf16* Vtb = Kbf + nqkv; // stored [bh][D][L]
  (void)ws_size; (void)in_sizes; (void)n_in; (void)out_size;

  dim3 pgrid(L_ / PROJ_ROWS, B_ * H_);
  proj_kernel<<<pgrid, 256, 0, stream>>>(query, Wq, bq, Qbf, 0);
  proj_kernel<<<pgrid, 256, 0, stream>>>(key,   Wk, bk, Kbf, 0);
  proj_kernel<<<pgrid, 256, 0, stream>>>(value, Wv, bv, Vtb, 1);

  dim3 agrid(L_ / QT, B_ * H_);
  attn_kernel<<<agrid, 256, 0, stream>>>(Qbf, Kbf, Vtb, masks, out);
}
